// LSTM_19018115186880
// MI455X (gfx1250) — matrix operand order repaired and checked by reading, not compiled, not run on hardware
//
#include <hip/hip_runtime.h>

// ---------------------------------------------------------------------------
// 2-layer LSTM (B=64, T=512, D=256, H=1024) + linear head, for gfx1250.
// All GEMM work on v_wmma_f32_16x16x32_bf16 (fp32 accumulate).
// Recurrent step fuses h@Whh^T + x_t@Wih^T + bias + gate nonlinearities.
// ---------------------------------------------------------------------------

typedef __bf16 bf16_t;
typedef bf16_t v16bf __attribute__((ext_vector_type(16)));
typedef float  v8f   __attribute__((ext_vector_type(8)));

#define B_   64
#define T_   512
#define D_   256
#define H_   1024
#define G4H_ 4096

static __device__ __forceinline__ unsigned short f2bf(float x) {
    unsigned int u = __float_as_uint(x);
    unsigned int r = (u + 0x7fffu + ((u >> 16) & 1u)) >> 16;   // round-nearest-even
    return (unsigned short)r;
}
static __device__ __forceinline__ float bf2f(unsigned short s) {
    return __uint_as_float(((unsigned int)s) << 16);
}

union Frag32 { v16bf v; uint4 u[2]; };

// A fragment: 16x32 bf16 tile from row-major [*, lda]; ISA layout:
// lane L (m=L&15, kh=L>>4): VGPR0-3 = K[k+kh*8 .. +7], VGPR4-7 = K[k+16+kh*8 .. +7]
static __device__ __forceinline__ v16bf load_a(const unsigned short* base, size_t lda,
                                               int k, int lane) {
    const int m = lane & 15, kh = lane >> 4;
    const unsigned short* p = base + (size_t)m * lda + (size_t)(k + kh * 8);
    Frag32 f;
    f.u[0] = *(const uint4*)p;
    f.u[1] = *(const uint4*)(p + 16);
    return f.v;
}

// B fragment: 32x16 bf16 tile == 16 rows of row-major W[N,K];
// lane L (n=L&15, kh=L>>4): VGPR0-3 = K[k+kh*8 .. +7], VGPR4-7 = K[k+16+kh*8 .. +7]
static __device__ __forceinline__ v16bf load_b(const unsigned short* wbase, size_t ldw,
                                               int n, int k, int lane) {
    const unsigned short* p = wbase + (size_t)(n + (lane & 15)) * ldw
                                    + (size_t)(k + (lane >> 4) * 8);
    Frag32 f;
    f.u[0] = *(const uint4*)p;
    f.u[1] = *(const uint4*)(p + 16);
    return f.v;
}

// ---------------------------------------------------------------------------
// One LSTM time step: gates = h_in@w_hh^T + x_t@w_ih^T + bias, then c/h update.
// grid = 32 blocks (32 H-columns each) x 256 threads (8 waves).
// wave w: gate q = w>>1, 16-col half = w&1; 4 M-tiles (all 64 batch rows).
// ---------------------------------------------------------------------------
__global__ __launch_bounds__(256) void lstm_step_kernel(
    const unsigned short* __restrict__ h_in,   // [B_, H_]  bf16
    const unsigned short* __restrict__ xin,    // [B_, T_, Din] bf16 (layer input seq)
    const unsigned short* __restrict__ w_hh,   // [G4H_, H_] bf16 row-major
    const unsigned short* __restrict__ w_ih,   // [G4H_, Din] bf16 row-major
    const float*          __restrict__ bias,   // [G4H_] (b_ih + b_hh)
    float*                __restrict__ cst,    // [B_, H_] f32 (in/out)
    unsigned short*       __restrict__ h_out,  // [B_, H_] bf16
    unsigned short*       __restrict__ hseq,   // [B_, T_, H_] bf16 or nullptr
    int t, int Din)
{
    __shared__ float gsm[4][B_][33];           // +1 pad vs 32-bank conflicts

    const int tid  = threadIdx.x;
    const int lane = tid & 31;
    const int w    = tid >> 5;
    const int q    = w >> 1;                   // gate 0..3 (i,f,g,o)
    const int j0   = blockIdx.x * 32;          // H-column block
    const int ncol = q * H_ + j0 + (w & 1) * 16;

    v8f acc[4];
#pragma unroll
    for (int m = 0; m < 4; ++m)
#pragma unroll
        for (int r = 0; r < 8; ++r) acc[m][r] = 0.0f;

    // ---- phase 1: h_in @ w_hh^T  (K = H_) ----
    for (int k = 0; k < H_; k += 32) {
        v16bf bfrag = load_b(w_hh, (size_t)H_, ncol, k, lane);
#pragma unroll
        for (int m = 0; m < 4; ++m) {
            v16bf afrag = load_a(h_in + (size_t)(m * 16) * H_, (size_t)H_, k, lane);
            acc[m] = __builtin_amdgcn_wmma_f32_16x16x32_bf16(
                false, afrag, false, bfrag, (short)0, acc[m], false, false);
        }
    }
    // ---- phase 2: x_t @ w_ih^T  (K = Din); x row b lives at (b*T_+t)*Din ----
    for (int k = 0; k < Din; k += 32) {
        v16bf bfrag = load_b(w_ih, (size_t)Din, ncol, k, lane);
#pragma unroll
        for (int m = 0; m < 4; ++m) {
            const unsigned short* abase = xin + ((size_t)(m * 16) * T_ + (size_t)t) * (size_t)Din;
            v16bf afrag = load_a(abase, (size_t)T_ * (size_t)Din, k, lane);
            acc[m] = __builtin_amdgcn_wmma_f32_16x16x32_bf16(
                false, afrag, false, bfrag, (short)0, acc[m], false, false);
        }
    }

    // ---- C layout: VGPR r, lanes 0-15 -> M=r, lanes 16-31 -> M=r+8 ----
    const int sub = (w & 1) * 16 + (lane & 15);
    const int mhi = 8 * (lane >> 4);
#pragma unroll
    for (int m = 0; m < 4; ++m)
#pragma unroll
        for (int r = 0; r < 8; ++r)
            gsm[q][m * 16 + r + mhi][sub] = acc[m][r];
    __syncthreads();

    // ---- elementwise gates over this block's 64x32 slice ----
    for (int e = tid; e < B_ * 32; e += 256) {
        const int b = e >> 5, jj = e & 31, j = j0 + jj;
        float ig = gsm[0][b][jj] + bias[0 * H_ + j];
        float fg = gsm[1][b][jj] + bias[1 * H_ + j];
        float gg = gsm[2][b][jj] + bias[2 * H_ + j];
        float og = gsm[3][b][jj] + bias[3 * H_ + j];
        ig = 1.0f / (1.0f + __expf(-ig));
        fg = 1.0f / (1.0f + __expf(-fg));
        og = 1.0f / (1.0f + __expf(-og));
        gg = tanhf(gg);
        const size_t idx = (size_t)b * H_ + (size_t)j;
        float c = fg * cst[idx] + ig * gg;
        float h = og * tanhf(c);
        cst[idx] = c;
        unsigned short hb = f2bf(h);
        h_out[idx] = hb;
        if (hseq) hseq[((size_t)b * T_ + (size_t)t) * H_ + (size_t)j] = hb;
    }
}

// ---------------------------------------------------------------------------
__global__ void f32_to_bf16_kernel(const float* __restrict__ in,
                                   unsigned short* __restrict__ out, int n) {
    int i = blockIdx.x * blockDim.x + threadIdx.x;
    if (i < n) out[i] = f2bf(in[i]);
}

__global__ void bias_sum_kernel(const float* __restrict__ a, const float* __restrict__ b,
                                float* __restrict__ out, int n) {
    int i = blockIdx.x * blockDim.x + threadIdx.x;
    if (i < n) out[i] = a[i] + b[i];
}

__global__ void zero_u32_kernel(unsigned int* __restrict__ p, int n) {
    int i = blockIdx.x * blockDim.x + threadIdx.x;
    if (i < n) p[i] = 0u;
}

// out[b] = b_lin + sum_j bf2f(h2[b,j]) * w_lin[j]   (O = 1)
__global__ __launch_bounds__(256) void final_linear_kernel(
    const unsigned short* __restrict__ h2, const float* __restrict__ w_lin,
    const float* __restrict__ b_lin, float* __restrict__ out)
{
    __shared__ float red[256];
    const int b = blockIdx.x;
    float s = 0.0f;
    for (int j = threadIdx.x; j < H_; j += 256)
        s += bf2f(h2[(size_t)b * H_ + j]) * w_lin[j];
    red[threadIdx.x] = s;
    __syncthreads();
    for (int off = 128; off > 0; off >>= 1) {
        if (threadIdx.x < off) red[threadIdx.x] += red[threadIdx.x + off];
        __syncthreads();
    }
    if (threadIdx.x == 0) out[b] = red[0] + b_lin[0];
}

// ---------------------------------------------------------------------------
extern "C" void kernel_launch(void* const* d_in, const int* in_sizes, int n_in,
                              void* d_out, int out_size, void* d_ws, size_t ws_size,
                              hipStream_t stream)
{
    (void)in_sizes; (void)n_in; (void)out_size; (void)ws_size;

    const float* x      = (const float*)d_in[0];
    const float* w_ih_0 = (const float*)d_in[1];
    const float* w_hh_0 = (const float*)d_in[2];
    const float* b_ih_0 = (const float*)d_in[3];
    const float* b_hh_0 = (const float*)d_in[4];
    const float* w_ih_1 = (const float*)d_in[5];
    const float* w_hh_1 = (const float*)d_in[6];
    const float* b_ih_1 = (const float*)d_in[7];
    const float* b_hh_1 = (const float*)d_in[8];
    const float* w_lin  = (const float*)d_in[9];
    const float* b_lin  = (const float*)d_in[10];
    float* out = (float*)d_out;

    // ---- workspace carve-up (~110 MB) ----
    char* ws = (char*)d_ws;
    auto carve = [&](size_t bytes) -> char* {
        char* p = ws; ws += (bytes + 255) & ~(size_t)255; return p;
    };
    unsigned short* x_bf   = (unsigned short*)carve((size_t)B_ * T_ * D_ * 2);
    unsigned short* h1seq  = (unsigned short*)carve((size_t)B_ * T_ * H_ * 2);
    unsigned short* wih0_b = (unsigned short*)carve((size_t)G4H_ * D_ * 2);
    unsigned short* whh0_b = (unsigned short*)carve((size_t)G4H_ * H_ * 2);
    unsigned short* wih1_b = (unsigned short*)carve((size_t)G4H_ * H_ * 2);
    unsigned short* whh1_b = (unsigned short*)carve((size_t)G4H_ * H_ * 2);
    float*          bias0  = (float*)carve((size_t)G4H_ * 4);
    float*          bias1  = (float*)carve((size_t)G4H_ * 4);
    float*          cst    = (float*)carve((size_t)B_ * H_ * 4);
    unsigned short* hbuf0  = (unsigned short*)carve((size_t)B_ * H_ * 2);
    unsigned short* hbuf1  = (unsigned short*)carve((size_t)B_ * H_ * 2);

    // ---- fp32 -> bf16 conversions & bias prep ----
    int n;
    n = B_ * T_ * D_;  f32_to_bf16_kernel<<<(n + 255) / 256, 256, 0, stream>>>(x,      x_bf,   n);
    n = G4H_ * D_;     f32_to_bf16_kernel<<<(n + 255) / 256, 256, 0, stream>>>(w_ih_0, wih0_b, n);
    n = G4H_ * H_;     f32_to_bf16_kernel<<<(n + 255) / 256, 256, 0, stream>>>(w_hh_0, whh0_b, n);
    n = G4H_ * H_;     f32_to_bf16_kernel<<<(n + 255) / 256, 256, 0, stream>>>(w_ih_1, wih1_b, n);
    n = G4H_ * H_;     f32_to_bf16_kernel<<<(n + 255) / 256, 256, 0, stream>>>(w_hh_1, whh1_b, n);
    n = G4H_;          bias_sum_kernel   <<<(n + 255) / 256, 256, 0, stream>>>(b_ih_0, b_hh_0, bias0, n);
    n = G4H_;          bias_sum_kernel   <<<(n + 255) / 256, 256, 0, stream>>>(b_ih_1, b_hh_1, bias1, n);

    // ---- layer 0 ----
    n = B_ * H_;       zero_u32_kernel<<<(n + 255) / 256, 256, 0, stream>>>((unsigned int*)cst,   n);
    n = B_ * H_ / 2;   zero_u32_kernel<<<(n + 255) / 256, 256, 0, stream>>>((unsigned int*)hbuf0, n);
    unsigned short* hin  = hbuf0;
    unsigned short* hout = hbuf1;
    for (int t = 0; t < T_; ++t) {
        lstm_step_kernel<<<32, 256, 0, stream>>>(hin, x_bf, whh0_b, wih0_b, bias0,
                                                 cst, hout, h1seq, t, D_);
        unsigned short* tmp = hin; hin = hout; hout = tmp;
    }

    // ---- layer 1 (input = h1seq; only final h needed) ----
    n = B_ * H_;       zero_u32_kernel<<<(n + 255) / 256, 256, 0, stream>>>((unsigned int*)cst,   n);
    n = B_ * H_ / 2;   zero_u32_kernel<<<(n + 255) / 256, 256, 0, stream>>>((unsigned int*)hbuf0, n);
    hin = hbuf0; hout = hbuf1;
    for (int t = 0; t < T_; ++t) {
        lstm_step_kernel<<<32, 256, 0, stream>>>(hin, h1seq, whh1_b, wih1_b, bias1,
                                                 cst, hout, (unsigned short*)nullptr, t, H_);
        unsigned short* tmp = hin; hin = hout; hout = tmp;
    }

    // ---- head: out[b] = h2[b,:] . w_lin + b_lin ----
    final_linear_kernel<<<B_, 256, 0, stream>>>(hin, w_lin, b_lin, out);
}
